// InfoNCELoss_80221399155344
// MI455X (gfx1250) — compile-verified, matrix operand order checked by reading, not run on hardware
//
#include <hip/hip_runtime.h>

typedef _Float16 v16h __attribute__((ext_vector_type(16)));
typedef _Float16 v8h  __attribute__((ext_vector_type(8)));
typedef float    v8f  __attribute__((ext_vector_type(8)));

#define B_ROWS   4096
#define NROWS    8192
#define DDIM     768
#define KSTEPS   (DDIM / 32)
#define INV_T    10.0f
#define SHIFT    10.0f                  // fixed logsumexp shift (|logit| <= ~10)
#define LDPAD    8
#define LDSTRIDE (DDIM + LDPAD)         // 776 halves = 1552 B; cols hit distinct banks
#define K2E      14.42695040888963f     // 10 * log2(e):  exp(10*acc - 10) = exp2(acc*K2E - K2E)
#define L2E      1.4426950408889634f    // log2(e)

// ---------------------------------------------------------------------------
// Kernel 1: L2-normalize rows of concat(z1, z2), store as f16 (L2-resident)
// ---------------------------------------------------------------------------
__global__ __launch_bounds__(256) void normalize_kernel(const float* __restrict__ z1,
                                                        const float* __restrict__ z2,
                                                        _Float16* __restrict__ fnh) {
    __shared__ float red[256];
    const int r = blockIdx.x;
    const float* src = (r < B_ROWS) ? (z1 + (size_t)r * DDIM)
                                    : (z2 + (size_t)(r - B_ROWS) * DDIM);
    float x[3];
    float ss = 0.0f;
#pragma unroll
    for (int i = 0; i < 3; ++i) {
        x[i] = src[threadIdx.x + i * 256];
        ss += x[i] * x[i];
    }
    red[threadIdx.x] = ss;
    __syncthreads();
    for (int s = 128; s > 0; s >>= 1) {
        if (threadIdx.x < s) red[threadIdx.x] += red[threadIdx.x + s];
        __syncthreads();
    }
    const float inv = 1.0f / fmaxf(sqrtf(red[0]), 1e-8f);
#pragma unroll
    for (int i = 0; i < 3; ++i)
        fnh[(size_t)r * DDIM + threadIdx.x + i * 256] = (_Float16)(x[i] * inv);
}

// ---------------------------------------------------------------------------
// Kernel 2: fused sim = (fn fn^T)/T with fixed-shift sum-of-exp + positive
// extraction. 8 waves/block; each wave owns a 32-row x 32-col tile per
// iteration: 2 A strips x 2 B strips -> 4 accumulators, 4 WMMAs per
// (4 global + 4 ds) b128 loads. Block shares a 32-column B tile in LDS.
// Core op: v_wmma_f32_16x16x32_f16.
// ---------------------------------------------------------------------------
__global__ __launch_bounds__(256) void fused_sim_lse_kernel(const _Float16* __restrict__ fnh,
                                                            float* __restrict__ row_loss) {
    __shared__ __align__(16) _Float16 Btile[32 * LDSTRIDE];   // 48.5 KB

    const int tid  = threadIdx.x;
    const int lane = tid & 31;
    const int wave = tid >> 5;
    const int hfl  = lane >> 4;     // 0: lanes 0-15, 1: lanes 16-31
    const int ln16 = lane & 15;
    const int rowbase  = (blockIdx.x * 8 + wave) * 32;              // 32-aligned
    const int postile  = (rowbase + B_ROWS) & (NROWS - 1);          // 32-aligned
    const int diagtile = rowbase;

    // A operands: lane holds row (strip + ln16); ISA 7.12.2 16-bit A layout:
    // halves 0..7 -> K = k0 + hfl*8 .. ; halves 8..15 -> K = k0 + 16 + hfl*8 ..
    const _Float16* aptr0 = fnh + (size_t)(rowbase + ln16) * DDIM + hfl * 8;
    const _Float16* aptr1 = aptr0 + (size_t)16 * DDIM;

    float rs0[8], pa0[8], rs1[8], pa1[8];
#pragma unroll
    for (int v = 0; v < 8; ++v) { rs0[v] = 0.0f; pa0[v] = 0.0f; rs1[v] = 0.0f; pa1[v] = 0.0f; }

    for (int su = 0; su < NROWS / 32; ++su) {
        const int colbase = su * 32;

        __syncthreads();
        // Stage B tile (32 cols x 768 halves) into LDS: 3072 chunks of 8
        // halves, 12 per thread. Prefetch next supertile toward the WGP.
#pragma unroll
        for (int t = 0; t < 12; ++t) {
            const int chunk = tid + t * 256;      // 0..3071
            const int c = chunk / 96;             // column within tile
            const int k = (chunk % 96) * 8;       // K offset (halves)
            const _Float16* g = fnh + (size_t)(colbase + c) * DDIM + k;
            *(float4*)(&Btile[c * LDSTRIDE + k]) = *(const float4*)g;
            if (su + 1 < NROWS / 32)
                __builtin_prefetch(g + (size_t)32 * DDIM, 0, 0);
        }
        __syncthreads();

        v8f acc00 = {}, acc01 = {}, acc10 = {}, acc11 = {};
        const _Float16* b0 = &Btile[ln16 * LDSTRIDE + hfl * 8];
        const _Float16* b1 = b0 + 16 * LDSTRIDE;

#pragma unroll 2
        for (int ks = 0; ks < KSTEPS; ++ks) {
            const int k0 = ks * 32;
            v8h a0lo = *(const v8h*)(aptr0 + k0);
            v8h a0hi = *(const v8h*)(aptr0 + k0 + 16);
            v16h a0 = __builtin_shufflevector(a0lo, a0hi, 0,1,2,3,4,5,6,7,8,9,10,11,12,13,14,15);
            v8h a1lo = *(const v8h*)(aptr1 + k0);
            v8h a1hi = *(const v8h*)(aptr1 + k0 + 16);
            v16h a1 = __builtin_shufflevector(a1lo, a1hi, 0,1,2,3,4,5,6,7,8,9,10,11,12,13,14,15);
            v8h b0lo = *(const v8h*)(b0 + k0);
            v8h b0hi = *(const v8h*)(b0 + k0 + 16);
            v16h bb0 = __builtin_shufflevector(b0lo, b0hi, 0,1,2,3,4,5,6,7,8,9,10,11,12,13,14,15);
            v8h b1lo = *(const v8h*)(b1 + k0);
            v8h b1hi = *(const v8h*)(b1 + k0 + 16);
            v16h bb1 = __builtin_shufflevector(b1lo, b1hi, 0,1,2,3,4,5,6,7,8,9,10,11,12,13,14,15);
            acc00 = __builtin_amdgcn_wmma_f32_16x16x32_f16(false, a0, false, bb0,
                                                           (short)0, acc00, false, false);
            acc01 = __builtin_amdgcn_wmma_f32_16x16x32_f16(false, a0, false, bb1,
                                                           (short)0, acc01, false, false);
            acc10 = __builtin_amdgcn_wmma_f32_16x16x32_f16(false, a1, false, bb0,
                                                           (short)0, acc10, false, false);
            acc11 = __builtin_amdgcn_wmma_f32_16x16x32_f16(false, a1, false, bb1,
                                                           (short)0, acc11, false, false);
        }

        // Epilogue. Wave-uniform scalar branch: only the diagonal supertile
        // and the positive supertile need per-element checks.
        if (colbase != diagtile && colbase != postile) {
            // Fast path: rs += exp(10*acc - 10), one fma + exp2 + add per elem.
#pragma unroll
            for (int v = 0; v < 8; ++v) {
                rs0[v] += exp2f(fmaf(acc00[v], K2E, -K2E));
                rs0[v] += exp2f(fmaf(acc01[v], K2E, -K2E));
                rs1[v] += exp2f(fmaf(acc10[v], K2E, -K2E));
                rs1[v] += exp2f(fmaf(acc11[v], K2E, -K2E));
            }
        } else {
#pragma unroll
            for (int v = 0; v < 8; ++v) {
                const int r0 = rowbase + hfl * 8 + v;       // strip 0 row
                const int r1 = r0 + 16;                     // strip 1 row
                const int p0 = (r0 + B_ROWS) & (NROWS - 1);
                const int p1 = (r1 + B_ROWS) & (NROWS - 1);
                const int c0 = colbase + ln16;
                const int c1 = colbase + 16 + ln16;
                const float s00 = acc00[v] * INV_T;
                const float s01 = acc01[v] * INV_T;
                const float s10 = acc10[v] * INV_T;
                const float s11 = acc11[v] * INV_T;
                if (c0 == p0) pa0[v] += s00;
                if (c1 == p0) pa0[v] += s01;
                if (c0 == p1) pa1[v] += s10;
                if (c1 == p1) pa1[v] += s11;
                if (c0 != r0) rs0[v] += exp2f((s00 - SHIFT) * L2E);
                if (c1 != r0) rs0[v] += exp2f((s01 - SHIFT) * L2E);
                if (c0 != r1) rs1[v] += exp2f((s10 - SHIFT) * L2E);
                if (c1 != r1) rs1[v] += exp2f((s11 - SHIFT) * L2E);
            }
        }
    }

    // Sum the 16 per-lane partials of each row (xor offsets 1,2,4,8 stay
    // inside each 16-lane half of the wave32).
#pragma unroll
    for (int v = 0; v < 8; ++v) {
        float s0 = rs0[v], p0 = pa0[v], s1 = rs1[v], p1 = pa1[v];
#pragma unroll
        for (int off = 1; off < 16; off <<= 1) {
            s0 += __shfl_xor(s0, off, 32);
            p0 += __shfl_xor(p0, off, 32);
            s1 += __shfl_xor(s1, off, 32);
            p1 += __shfl_xor(p1, off, 32);
        }
        rs0[v] = s0; pa0[v] = p0; rs1[v] = s1; pa1[v] = p1;
    }

    if (ln16 == 0) {   // lanes 0/16 hold rows base+0..7 / base+8..15 per strip
        const int base = rowbase + hfl * 8;
#pragma unroll
        for (int v = 0; v < 8; ++v) {
            row_loss[base + v]      = (SHIFT + __logf(rs0[v])) - pa0[v];
            row_loss[base + 16 + v] = (SHIFT + __logf(rs1[v])) - pa1[v];
        }
    }
}

// ---------------------------------------------------------------------------
// Kernel 3: mean over the 8192 per-row losses -> scalar
// ---------------------------------------------------------------------------
__global__ __launch_bounds__(256) void mean_kernel(const float* __restrict__ row_loss,
                                                   float* __restrict__ out) {
    __shared__ float red[256];
    float s = 0.0f;
    for (int i = threadIdx.x; i < NROWS; i += 256) s += row_loss[i];
    red[threadIdx.x] = s;
    __syncthreads();
    for (int k = 128; k > 0; k >>= 1) {
        if (threadIdx.x < k) red[threadIdx.x] += red[threadIdx.x + k];
        __syncthreads();
    }
    if (threadIdx.x == 0) out[0] = red[0] / (float)NROWS;
}

extern "C" void kernel_launch(void* const* d_in, const int* in_sizes, int n_in,
                              void* d_out, int out_size, void* d_ws, size_t ws_size,
                              hipStream_t stream) {
    const float* z1 = (const float*)d_in[0];
    const float* z2 = (const float*)d_in[1];

    _Float16* fnh   = (_Float16*)d_ws;                                   // 12.6 MB
    float* row_loss = (float*)((char*)d_ws +
                               (size_t)NROWS * DDIM * sizeof(_Float16)); // +32 KB

    normalize_kernel<<<NROWS, 256, 0, stream>>>(z1, z2, fnh);
    fused_sim_lse_kernel<<<NROWS / (32 * 8), 256, 0, stream>>>(fnh, row_loss);
    mean_kernel<<<1, 256, 0, stream>>>(row_loss, (float*)d_out);
}
